// Izhikevich_SNN_80247168958996
// MI455X (gfx1250) — matrix operand order repaired and checked by reading, not compiled, not run on hardware
//
#include <hip/hip_runtime.h>

// ---------------------------------------------------------------------------
// Izhikevich SNN, restructured for MI455X (gfx1250):
//   Phase 1: I[t,b,n] = X[b,t,:] @ W1[n,:]  as one 64000x512x512 bf16 WMMA GEMM
//            (I stashed into the u-plane of the states output region).
//            GEMM row m == (t,b) = (m>>6, m&63); X is [B,T,N] so the row's
//            source address is (m&63)*T*N + (m>>6)*N  (rows stay contiguous).
//   Phase 2: barrier-free per-(b,n) recurrence over t, fused wave-reduce dot
//            with W2 accumulated atomically into decoded[t,b]
//   Phase 3: sequential leak scan over decoded (tiny)
// Grid for phase 1 is (4 N-tiles, 500 M-tiles): x-fastest launch order keeps
// the 4 consumers of each 256KB X tile adjacent -> X read from HBM once
// (L2 = 192MB), W1 (1MB) always L2-resident.
// ---------------------------------------------------------------------------

#define TT 1000
#define BB 64
#define NN 512
#define KK 512
#define BN (BB * NN)                       // 32768
static constexpr size_t TBNs = (size_t)TT * BN;   // 32,768,000
static constexpr size_t TNs  = (size_t)TT * NN;   // 512,000 (X batch stride)

typedef __attribute__((ext_vector_type(16))) __bf16 bf16x16;
typedef __attribute__((ext_vector_type(8)))  __bf16 bf16x8;
typedef __attribute__((ext_vector_type(8)))  float  v8f;

union FragU { bf16x16 v; bf16x8 h[2]; };

// X source offset for GEMM row m (m = t*64 + b, X laid out [B,T,N])
__device__ __forceinline__ size_t x_row_off(int m) {
  return (size_t)(m & 63) * TNs + (size_t)(m >> 6) * NN;
}

// ---------------------------------------------------------------------------
// Phase 0: zero decoded region (accumulated via atomics each call)
// ---------------------------------------------------------------------------
__global__ void izh_zero_dec(float* __restrict__ p) {
  p[blockIdx.x * 256 + threadIdx.x] = 0.0f;
}

// ---------------------------------------------------------------------------
// Phase 1: WMMA GEMM  I = X @ W1^T
//   grid = (4, 500); block = 256 threads = 8 waves
//   workgroup tile: 128(M) x 128(N); wave tile: 32 x 64; K staged in 64-chunks
// ---------------------------------------------------------------------------
__global__ __launch_bounds__(256) void izh_gemm_wmma(
    const float* __restrict__ X, const float* __restrict__ W1,
    float* __restrict__ out)
{
  // rows padded to 72 bf16 (144B): keeps 16B alignment, spreads banks
  __shared__ __bf16 lA[128 * 72];
  __shared__ __bf16 lB[128 * 72];

  const int n0 = blockIdx.x * 128;   // N tile (4)  -- x-fastest for L2 reuse of X
  const int m0 = blockIdx.y * 128;   // M tile (500)
  const int t    = threadIdx.x;
  const int lane = t & 31;
  const int wave = t >> 5;
  const int r0 = (wave >> 1) * 32;   // wave M sub-tile
  const int c0 = (wave & 1) * 64;    // wave N sub-tile
  const int frow = lane & 15;
  const int kh   = lane >> 4;

  const int lr = t >> 3;             // 0..31  (loader row within pass)
  const int lc = (t & 7) * 8;        // 0..56  (loader col, 8 floats)

  v8f acc[2][4] = {};

  for (int kt = 0; kt < 8; ++kt) {
    const int k0 = kt * 64;
    if (kt) __syncthreads();         // previous compute done before LDS overwrite

    if (kt + 1 < 8) {                // global_prefetch_b8 for next K-chunk
      __builtin_prefetch(X  + x_row_off(m0 + lr) + k0 + 64 + lc, 0, 1);
      __builtin_prefetch(W1 + (size_t)(n0 + lr) * KK + k0 + 64 + lc, 0, 1);
    }

#pragma unroll
    for (int p = 0; p < 4; ++p) {
      const int row = p * 32 + lr;
      const float* xrow = X + x_row_off(m0 + row) + k0 + lc;
      const float4 fa0 = *(const float4*)(xrow);
      const float4 fa1 = *(const float4*)(xrow + 4);
      const float4 fb0 = *(const float4*)(W1 + (size_t)(n0 + row) * KK + k0 + lc);
      const float4 fb1 = *(const float4*)(W1 + (size_t)(n0 + row) * KK + k0 + lc + 4);
      bf16x8 ha, hb;
      ha[0] = (__bf16)fa0.x; ha[1] = (__bf16)fa0.y; ha[2] = (__bf16)fa0.z; ha[3] = (__bf16)fa0.w;
      ha[4] = (__bf16)fa1.x; ha[5] = (__bf16)fa1.y; ha[6] = (__bf16)fa1.z; ha[7] = (__bf16)fa1.w;
      hb[0] = (__bf16)fb0.x; hb[1] = (__bf16)fb0.y; hb[2] = (__bf16)fb0.z; hb[3] = (__bf16)fb0.w;
      hb[4] = (__bf16)fb1.x; hb[5] = (__bf16)fb1.y; hb[6] = (__bf16)fb1.z; hb[7] = (__bf16)fb1.w;
      *(bf16x8*)&lA[row * 72 + lc] = ha;
      *(bf16x8*)&lB[row * 72 + lc] = hb;
    }
    __syncthreads();

#pragma unroll
    for (int s = 0; s < 2; ++s) {    // two K=32 WMMA steps per chunk
      const int kk = s * 32;
      FragU aF[2], bF[4];
      // A 16x32 bf16 layout (ISA 7.12.2): lane half kh -> K = 8*kh..8*kh+7, 16+8*kh..
#pragma unroll
      for (int i = 0; i < 2; ++i) {
        aF[i].h[0] = *(const bf16x8*)&lA[(r0 + 16 * i + frow) * 72 + kk + kh * 8];
        aF[i].h[1] = *(const bf16x8*)&lA[(r0 + 16 * i + frow) * 72 + kk + 16 + kh * 8];
      }
      // B 32x16 bf16: lane = column n, half kh -> K = 8*kh..8*kh+7, 16+8*kh..
#pragma unroll
      for (int j = 0; j < 4; ++j) {
        bF[j].h[0] = *(const bf16x8*)&lB[(c0 + 16 * j + frow) * 72 + kk + kh * 8];
        bF[j].h[1] = *(const bf16x8*)&lB[(c0 + 16 * j + frow) * 72 + kk + 16 + kh * 8];
      }
#pragma unroll
      for (int i = 0; i < 2; ++i)
#pragma unroll
        for (int j = 0; j < 4; ++j)
          acc[i][j] = __builtin_amdgcn_wmma_f32_16x16x32_bf16(
              false, aF[i].v, false, bF[j].v, (short)0, acc[i][j], false, false);
    }
  }

  // Store I into the u-plane of states: out[TBN + t*3BN + b*N + n], m = t*64+b
  float* Iplane = out + TBNs;
  const int col16 = lane & 15;
  const int radd  = (lane >> 4) * 8;
#pragma unroll
  for (int i = 0; i < 2; ++i) {
#pragma unroll
    for (int j = 0; j < 4; ++j) {
      const int col = n0 + c0 + 16 * j + col16;
#pragma unroll
      for (int r = 0; r < 8; ++r) {
        const int m = m0 + r0 + 16 * i + r + radd;
        const size_t off = (size_t)(m >> 6) * (3 * BN) + (size_t)(m & 63) * NN + col;
        Iplane[off] = acc[i][j][r];
      }
    }
  }
}

// ---------------------------------------------------------------------------
// Phase 2: barrier-free recurrence, one thread per (b,n); fused W2 dot via
// wave-level shfl_xor reduce + one atomicAdd per wave per step.
// ---------------------------------------------------------------------------
__global__ __launch_bounds__(256) void izh_scan(
    float* __restrict__ out, const float* __restrict__ st,
    const float* __restrict__ W2,
    const float* __restrict__ pa,  const float* __restrict__ pb,
    const float* __restrict__ pc,  const float* __restrict__ pd,
    const float* __restrict__ pv2, const float* __restrict__ pv1,
    const float* __restrict__ pv0, const float* __restrict__ ptau,
    const float* __restrict__ pth, const float* __restrict__ pdt)
{
  const int idx = blockIdx.x * 256 + threadIdx.x;   // 0..32767 = b*512+n
  const int n = idx & (NN - 1);
  const int b = idx >> 9;

  float u = st[idx];
  float v = st[BN + idx];
  const float an = pa[n],  bn = pb[n],  cn = pc[n], dn = pd[n];
  const float q2 = pv2[n], q1 = pv1[n], q0 = pv0[n];
  const float dt = pdt[n], th = pth[n];
  const float itau = dt / ptau[n];
  const float w2 = W2[n];

  float* states = out + TBNs;
  float* dec    = out + 4 * TBNs;

#pragma unroll 1
  for (int ts = 0; ts < TT; ++ts) {
    float* sp = states + (size_t)ts * (3 * BN);
    const float I = sp[idx];                        // precomputed feedforward
    float vn = v + dt * (q2 * v * v + q1 * v + q0 - u + I);
    float un = u + itau * (an * (bn * v - u));
    const float s = vn > th ? 1.0f : 0.0f;
    vn = (s > 0.0f) ? cn : vn;
    un += s * dn;
    __builtin_nontemporal_store(s,  out + (size_t)ts * BN + idx);  // spikes
    __builtin_nontemporal_store(un, sp + idx);                     // u (overwrites I)
    __builtin_nontemporal_store(vn, sp + BN + idx);                // v
    __builtin_nontemporal_store(s,  sp + 2 * BN + idx);            // s
    u = un; v = vn;

    float p = s * w2;                               // wave32 reduce (same b per wave)
#pragma unroll
    for (int o = 16; o > 0; o >>= 1) p += __shfl_xor(p, o, 32);
    if ((threadIdx.x & 31) == 0) atomicAdd(&dec[ts * BB + b], p);
  }
}

// ---------------------------------------------------------------------------
// Phase 3: leak scan over decoded, in place. 64 threads, T=1000 sequential.
// ---------------------------------------------------------------------------
__global__ void izh_li(float* __restrict__ dec, const float* __restrict__ li0,
                       const float* __restrict__ leak)
{
  const int b = threadIdx.x;                        // 0..63
  float li = li0[b];
  const float lk = leak[0];
  for (int ts = 0; ts < TT; ++ts) {
    li = lk * li + dec[ts * BB + b];
    dec[ts * BB + b] = li;
  }
}

// ---------------------------------------------------------------------------
extern "C" void kernel_launch(void* const* d_in, const int* in_sizes, int n_in,
                              void* d_out, int out_size, void* d_ws, size_t ws_size,
                              hipStream_t stream) {
  const float* X    = (const float*)d_in[0];   // [64,1000,512]
  const float* st   = (const float*)d_in[1];   // [3,64,512]
  const float* li0  = (const float*)d_in[2];   // [64,1]
  const float* W1   = (const float*)d_in[3];   // [512,512]
  const float* W2   = (const float*)d_in[4];   // [1,512]
  const float* pa   = (const float*)d_in[5];
  const float* pb   = (const float*)d_in[6];
  const float* pc   = (const float*)d_in[7];
  const float* pd   = (const float*)d_in[8];
  const float* pv2  = (const float*)d_in[9];
  const float* pv1  = (const float*)d_in[10];
  const float* pv0  = (const float*)d_in[11];
  const float* ptau = (const float*)d_in[12];
  const float* pth  = (const float*)d_in[13];
  const float* plk  = (const float*)d_in[14];
  const float* pdt  = (const float*)d_in[15];
  float* out = (float*)d_out;

  izh_zero_dec<<<250, 256, 0, stream>>>(out + 4 * TBNs);
  izh_gemm_wmma<<<dim3(4, 500), 256, 0, stream>>>(X, W1, out);
  izh_scan<<<128, 256, 0, stream>>>(out, st, W2, pa, pb, pc, pd,
                                    pv2, pv1, pv0, ptau, pth, pdt);
  izh_li<<<1, 64, 0, stream>>>(out + 4 * TBNs, li0, plk);
}
